// PlatonicConv_26001732010285
// MI455X (gfx1250) — matrix operand order repaired and checked by reading, not compiled, not run on hardware
//
#include <hip/hip_runtime.h>
#include <math.h>

// ---------------- problem constants ----------------
#define NN   32768
#define CC   384
#define EE   384
#define GG   12
#define DD   32
#define PP   16
#define NGR  64
#define INV_AVG (1.0f / 512.0f)

typedef __attribute__((ext_vector_type(16))) __bf16 v16bf;
typedef __attribute__((ext_vector_type(8)))  float  v8f;

#if defined(__has_builtin)
#if __has_builtin(__builtin_amdgcn_global_load_async_to_lds_b128) && \
    __has_builtin(__builtin_amdgcn_s_wait_asynccnt)
#define HAVE_ASYNC_LDS 1
#endif
#endif

#if HAVE_ASYNC_LDS
// Builtin expects: (v4i32 addrspace(1)*, v4i32 addrspace(3)*, imm offset, imm cpol)
typedef int v4i_vs __attribute__((vector_size(16)));
typedef __attribute__((address_space(1))) v4i_vs v4i_g;
typedef __attribute__((address_space(3))) v4i_vs v4i_l;
#endif

__device__ __forceinline__ v8f wmma_bf16(v16bf a, v16bf b, v8f c) {
  // 8 args: (neg_a, A, neg_b, B, c_mod, C, reuse_a, reuse_b)
  return __builtin_amdgcn_wmma_f32_16x16x32_bf16(false, a, false, b, (short)0, c, false, false);
}

// ---------------------------------------------------------------------------
// Pre-swizzle a f32 [K,Nn] weight matrix into bf16 WMMA B-operand order:
//   WS[((ktile*(Nn/16) + ntile)*32 + lane)*16 + i] = (bf16)W[ktile*32 + i +
//   (lane>>4)*16][ntile*16 + (lane&15)]
// so a wave's B operand is a contiguous 32B chunk per lane.
// ---------------------------------------------------------------------------
__global__ void prep_weights(const float* __restrict__ W,
                             __bf16* __restrict__ WS, int K, int Nn) {
  const int nnt = Nn / 16;
  const int total = (K / 32) * nnt * 32;
  const int t = blockIdx.x * blockDim.x + threadIdx.x;
  if (t >= total) return;
  const int lane  = t & 31;
  const int tt    = t >> 5;
  const int ntile = tt % nnt;
  const int ktile = tt / nnt;
  const int half  = lane >> 4;
  const int l16   = lane & 15;
  const int col   = ntile * 16 + l16;
  __bf16* dst = WS + (size_t)t * 16;
#pragma unroll
  for (int i = 0; i < 16; ++i) {
    const int k = ktile * 32 + i + half * 16;      // B k-map
    dst[i] = (__bf16)W[k * Nn + col];
  }
}

// ---------------------------------------------------------------------------
// GEMM: C[M,Nn] = A[M,K](f32) * W(pre-swizzled bf16) + bias.
// Block = 128 threads (4 waves), block tile 64(M) x 64(N).
// Each wave owns 16 rows and all four 16-col subtiles (A reused x4).
// B operands staged global->LDS (async path when available).
// ---------------------------------------------------------------------------
__global__ void gemm_bf16wmma(const float* __restrict__ A,
                              const __bf16* __restrict__ WS,
                              const float* __restrict__ bias,
                              float* __restrict__ C,
                              int M, int K, int Nn) {
  const int tid  = threadIdx.x;        // 0..127
  const int wave = tid >> 5;           // 0..3
  const int lane = tid & 31;
  const int half = lane >> 4;
  const int l16  = lane & 15;
  const int nnt  = Nn / 16;

  __shared__ __align__(32) __bf16 Bs[4 * 32 * 16];   // 4KB: 4 ntiles x 32 lanes x 16 halfs

  const int rowA = blockIdx.x * 64 + wave * 16 + l16;
  const float* arow = A + (size_t)rowA * K;

  v8f acc[4];
#pragma unroll
  for (int nt = 0; nt < 4; ++nt) {
    const int col = blockIdx.y * 64 + nt * 16 + l16;
    const float bval = bias ? bias[col] : 0.0f;
#pragma unroll
    for (int r = 0; r < 8; ++r) acc[nt][r] = bval;
  }

  const int nkt = K / 32;
  for (int ktile = 0; ktile < nkt; ++ktile) {
    const int k0 = ktile * 32;

    // ---- stage 4KB of pre-swizzled B for this (ktile, 64-col block) ----
    const __bf16* gB = WS + ((size_t)(ktile * nnt + blockIdx.y * 4) * 32) * 16;
#if HAVE_ASYNC_LDS
    __builtin_amdgcn_global_load_async_to_lds_b128(
        (v4i_g*)(gB + tid * 16), (v4i_l*)(Bs + tid * 16), 0, 0);
    __builtin_amdgcn_global_load_async_to_lds_b128(
        (v4i_g*)(gB + tid * 16 + 8), (v4i_l*)(Bs + tid * 16 + 8), 0, 0);
    __builtin_amdgcn_s_wait_asynccnt(0);
#else
    ((uint4*)Bs)[tid * 2]     = ((const uint4*)gB)[tid * 2];
    ((uint4*)Bs)[tid * 2 + 1] = ((const uint4*)gB)[tid * 2 + 1];
#endif
    __syncthreads();

    // ---- A operand: 16 consecutive k per lane (A k-map) ----
    v16bf a;
#pragma unroll
    for (int j = 0; j < 8; ++j) {
      a[j]     = (__bf16)arow[k0 + half * 16 + j];
      a[j + 8] = (__bf16)arow[k0 + half * 16 + 8 + j];
    }
    if (k0 + 32 < K) __builtin_prefetch(arow + k0 + 32, 0, 0);

    const v16bf* bsv = (const v16bf*)Bs;
#pragma unroll
    for (int nt = 0; nt < 4; ++nt) {
      const v16bf b = bsv[nt * 32 + lane];
      acc[nt] = wmma_bf16(a, b, acc[nt]);
    }
    __syncthreads();
  }

#pragma unroll
  for (int nt = 0; nt < 4; ++nt) {
    const int col = blockIdx.y * 64 + nt * 16 + l16;
#pragma unroll
    for (int r = 0; r < 8; ++r) {
      const int row = blockIdx.x * 64 + wave * 16 + half * 8 + r;   // C/D m-map
      C[(size_t)row * Nn + col] = acc[nt][r];
    }
  }
}

// ---------------------------------------------------------------------------
// RoPE: phase[n,g,p] = sum_s pos[n,s] * freqs[g,0,p,s]; rotate Q pairs
// in-place; K pairs are (cos - sin, cos + sin) since k_raw = ones.
// ---------------------------------------------------------------------------
__global__ void rope_kernel(const float* __restrict__ pos,
                            const float* __restrict__ freqs,
                            float* __restrict__ Q,      // [N,384] in-place
                            float* __restrict__ Karr) { // [N,384]
  const int idx = blockIdx.x * blockDim.x + threadIdx.x;
  if (idx >= NN * GG * PP) return;
  const int p = idx % PP;
  const int t = idx / PP;
  const int g = t % GG;
  const int n = t / GG;

  float ph = 0.0f;
#pragma unroll
  for (int s = 0; s < 3; ++s)
    ph += pos[n * 3 + s] * freqs[(g * PP + p) * 3 + s];

  const float c = cosf(ph);
  const float s = sinf(ph);

  const int base = n * EE + g * DD + 2 * p;
  const float q1 = Q[base];
  const float q2 = Q[base + 1];
  Q[base]     = q1 * c - q2 * s;
  Q[base + 1] = q1 * s + q2 * c;
  Karr[base]     = c - s;
  Karr[base + 1] = c + s;
}

// ---------------------------------------------------------------------------
// Per-graph start offsets via binary search (batch is sorted).
// ---------------------------------------------------------------------------
__global__ void bounds_kernel(const int* __restrict__ batch,
                              int* __restrict__ starts) {
  const int j = threadIdx.x;
  if (j > NGR) return;
  int lo = 0, hi = NN;
  while (lo < hi) {
    const int mid = (lo + hi) >> 1;
    if (batch[mid] < j) lo = mid + 1; else hi = mid;
  }
  starts[j] = lo;
}

// ---------------------------------------------------------------------------
// kv_kernel[j,g,d,e] = (1/512) * sum_{n in graph j} k[n,g,d]*v[n,g,e]
//  == (Kseg^T * Vseg)/512, one wave per (graph, g), 2x2 WMMA tiles.
// ---------------------------------------------------------------------------
__global__ void kv_kernel(const float* __restrict__ Karr,
                          const float* __restrict__ Varr,
                          const int* __restrict__ starts,
                          float* __restrict__ KVK) {
  const int j    = blockIdx.x;
  const int g    = blockIdx.y;
  const int lane = threadIdx.x;
  const int half = lane >> 4;
  const int l16  = lane & 15;
  const int s    = starts[j];
  const int e    = starts[j + 1];

  v8f c00 = {}, c01 = {}, c10 = {}, c11 = {};

  for (int n0 = s; n0 < e; n0 += 32) {
    v16bf a0, a1, b0, b1;
#pragma unroll
    for (int i = 0; i < 16; ++i) {
      const int nk = n0 + i + half * 16;  // A k-map
      float ka0 = 0.0f, ka1 = 0.0f;
      if (nk < e) {
        ka0 = Karr[nk * EE + g * DD + l16];
        ka1 = Karr[nk * EE + g * DD + 16 + l16];
      }
      a0[i] = (__bf16)ka0;
      a1[i] = (__bf16)ka1;
      const int nb = n0 + i + half * 16;                          // B k-map
      float vb0 = 0.0f, vb1 = 0.0f;
      if (nb < e) {
        vb0 = Varr[nb * EE + g * DD + l16];
        vb1 = Varr[nb * EE + g * DD + 16 + l16];
      }
      b0[i] = (__bf16)vb0;
      b1[i] = (__bf16)vb1;
    }
    c00 = wmma_bf16(a0, b0, c00);
    c01 = wmma_bf16(a0, b1, c01);
    c10 = wmma_bf16(a1, b0, c10);
    c11 = wmma_bf16(a1, b1, c11);
  }

  const int base = (j * GG + g) * DD;
#pragma unroll
  for (int r = 0; r < 8; ++r) {
    const int d = half * 8 + r;
    KVK[(base + d) * DD + l16]           = c00[r] * INV_AVG;
    KVK[(base + d) * DD + 16 + l16]      = c01[r] * INV_AVG;
    KVK[(base + 16 + d) * DD + l16]      = c10[r] * INV_AVG;
    KVK[(base + 16 + d) * DD + 16 + l16] = c11[r] * INV_AVG;
  }
}

// ---------------------------------------------------------------------------
// Mid[n, g*32+e] = sum_d Qrot[n,g,d] * KVK[graph(n),g,d,e]
// One wave per (graph, g); B (32x32) loaded once, 16-row node tiles.
// ---------------------------------------------------------------------------
__global__ void attn_out_kernel(const float* __restrict__ Qrot,
                                const float* __restrict__ KVK,
                                const int* __restrict__ starts,
                                float* __restrict__ Mid) {
  const int j    = blockIdx.x;
  const int g    = blockIdx.y;
  const int lane = threadIdx.x;
  const int half = lane >> 4;
  const int l16  = lane & 15;
  const int s    = starts[j];
  const int e    = starts[j + 1];

  const int base = (j * GG + g) * DD;
  v16bf b0, b1;
#pragma unroll
  for (int i = 0; i < 16; ++i) {
    const int d = i + half * 16;               // B k-map over d
    b0[i] = (__bf16)KVK[(base + d) * DD + l16];
    b1[i] = (__bf16)KVK[(base + d) * DD + 16 + l16];
  }

  for (int n0 = s; n0 < e; n0 += 16) {
    v16bf a;
    const int row = n0 + l16;                  // A m-map over nodes
#pragma unroll
    for (int i = 0; i < 16; ++i) {
      const int d = i + half * 16;  // A k-map over d
      a[i] = (__bf16)((row < e) ? Qrot[row * EE + g * DD + d] : 0.0f);
    }
    v8f z = {};
    const v8f c0 = wmma_bf16(a, b0, z);
    const v8f c1 = wmma_bf16(a, b1, z);
#pragma unroll
    for (int r = 0; r < 8; ++r) {
      const int rr = n0 + half * 8 + r;
      if (rr < e) {
        Mid[rr * EE + g * DD + l16]      = c0[r];
        Mid[rr * EE + g * DD + 16 + l16] = c1[r];
      }
    }
  }
}

// ---------------------------------------------------------------------------
extern "C" void kernel_launch(void* const* d_in, const int* in_sizes, int n_in,
                              void* d_out, int out_size, void* d_ws, size_t ws_size,
                              hipStream_t stream) {
  (void)in_sizes; (void)n_in; (void)out_size; (void)ws_size;

  const float* x     = (const float*)d_in[0];
  const float* pos   = (const float*)d_in[1];
  const int*   batch = (const int*)  d_in[2];
  const float* Wq    = (const float*)d_in[3];
  const float* bq    = (const float*)d_in[4];
  const float* Wv    = (const float*)d_in[5];
  const float* bv    = (const float*)d_in[6];
  const float* Wo    = (const float*)d_in[7];
  const float* bo    = (const float*)d_in[8];
  const float* freqs = (const float*)d_in[9];
  float* out = (float*)d_out;

  // ---- workspace carve-up (256B-aligned regions) ----
  const size_t NE      = (size_t)NN * EE;                 // 12,582,912 f32
  const size_t WS_ELTS = (size_t)(CC / 32) * (EE / 16) * 32 * 16;  // 147,456 bf16
  char* ws = (char*)d_ws;
  size_t off = 0;
  auto carve = [&](size_t bytes) {
    char* p = ws + off;
    off = (off + bytes + 255) & ~(size_t)255;
    return (void*)p;
  };
  float*  Q      = (float*)carve(NE * 4);                 // Q, rotated in-place
  float*  V      = (float*)carve(NE * 4);                 // V, reused as Mid
  float*  Karr   = (float*)carve(NE * 4);
  float*  KVK    = (float*)carve((size_t)NGR * GG * DD * DD * 4);
  __bf16* WSq    = (__bf16*)carve(WS_ELTS * 2);
  __bf16* WSv    = (__bf16*)carve(WS_ELTS * 2);
  __bf16* WSo    = (__bf16*)carve(WS_ELTS * 2);
  int*    starts = (int*)carve((NGR + 1) * 4);
  float*  Mid    = V;

  // 0) pre-swizzle the three weight matrices into bf16 operand order
  {
    const int total = (CC / 32) * (EE / 16) * 32;         // 9216 threads
    const int blks = (total + 255) / 256;
    prep_weights<<<blks, 256, 0, stream>>>(Wq, WSq, CC, EE);
    prep_weights<<<blks, 256, 0, stream>>>(Wv, WSv, CC, EE);
    prep_weights<<<blks, 256, 0, stream>>>(Wo, WSo, EE, CC);
  }

  const dim3 gemmGrid(NN / 64, EE / 64);                  // 512 x 6 blocks
  const dim3 gemmBlk(128);                                // 4 waves

  // 1) projections
  gemm_bf16wmma<<<gemmGrid, gemmBlk, 0, stream>>>(x, WSq, bq, Q, NN, CC, EE);
  gemm_bf16wmma<<<gemmGrid, gemmBlk, 0, stream>>>(x, WSv, bv, V, NN, CC, EE);

  // 2) RoPE rotate q in-place + build k
  {
    const int total = NN * GG * PP;
    rope_kernel<<<(total + 255) / 256, 256, 0, stream>>>(pos, freqs, Q, Karr);
  }

  // 3) per-graph segment bounds
  bounds_kernel<<<1, 128, 0, stream>>>(batch, starts);

  // 4) segment-sum of outer products as per-graph K^T V GEMM
  kv_kernel<<<dim3(NGR, GG), dim3(32), 0, stream>>>(Karr, V, starts, KVK);

  // 5) apply per-graph 32x32 kernel to rotated q (overwrites V as Mid)
  attn_out_kernel<<<dim3(NGR, GG), dim3(32), 0, stream>>>(Q, KVK, starts, Mid);

  // 6) output projection
  gemm_bf16wmma<<<gemmGrid, gemmBlk, 0, stream>>>(Mid, WSo, bo, out, NN, EE, CC);
}
